// TemporalAdditiveAttention_6347961663692
// MI455X (gfx1250) — matrix operand order repaired and checked by reading, not compiled, not run on hardware
//
#include <hip/hip_runtime.h>
#include <hip/hip_bf16.h>

#define BB 64
#define TT 512
#define HH 1024
#define ROWS (BB * TT)   // 32768

// Padded LDS row stride for the x tile: 1032 bf16 = 2064 B = 516 dwords.
// 516 mod 64 = 4 -> the 16 lanes of an A-frag ds_load_b128 hit disjoint
// 4-dword bank groups (conflict-free) instead of all aliasing to one bank set.
#define LDS_STRIDE 1032

typedef __bf16 bf16_t;
typedef bf16_t v16bf __attribute__((ext_vector_type(16)));
typedef bf16_t v8bf  __attribute__((ext_vector_type(8)));
typedef float  v8f   __attribute__((ext_vector_type(8)));

#if __has_builtin(__builtin_amdgcn_tanhf)
#define TANHF(x) __builtin_amdgcn_tanhf(x)
#else
#define TANHF(x) tanhf(x)
#endif

__device__ __forceinline__ unsigned short f2bf(float f) {
  unsigned int u = __float_as_uint(f);
  unsigned int r = u + 0x7FFFu + ((u >> 16) & 1u);   // round-to-nearest-even
  return (unsigned short)(r >> 16);
}
__device__ __forceinline__ unsigned int pack2bf(float a, float b) {
  return (unsigned int)f2bf(a) | ((unsigned int)f2bf(b) << 16);
}

// ---------------------------------------------------------------------------
// Kernel 1: LayerNorm, one wave32 per row. Writes x as f32 and bf16.
// ---------------------------------------------------------------------------
__global__ void __launch_bounds__(256)
ln_kernel(const float* __restrict__ xin, const float* __restrict__ gamma,
          const float* __restrict__ beta, float* __restrict__ xf,
          unsigned short* __restrict__ xbf) {
  const int wid  = threadIdx.x >> 5;
  const int lane = threadIdx.x & 31;
  const int row  = blockIdx.x * 8 + wid;

  const float4* src = (const float4*)(xin + (size_t)row * HH);
  float4 d[8];
  float s = 0.f, s2 = 0.f;
#pragma unroll
  for (int j = 0; j < 8; ++j) {
    d[j] = src[j * 32 + lane];
    s  += d[j].x + d[j].y + d[j].z + d[j].w;
    s2 += d[j].x * d[j].x + d[j].y * d[j].y + d[j].z * d[j].z + d[j].w * d[j].w;
  }
#pragma unroll
  for (int m = 1; m <= 16; m <<= 1) {
    s  += __shfl_xor(s,  m, 32);
    s2 += __shfl_xor(s2, m, 32);
  }
  const float mu   = s * (1.0f / HH);
  const float var  = fmaxf(s2 * (1.0f / HH) - mu * mu, 0.0f);
  const float rinv = rsqrtf(var + 1e-5f);

  float4* dstf = (float4*)(xf + (size_t)row * HH);
  uint2*  dstb = (uint2*)(xbf + (size_t)row * HH);
  const float4* g4 = (const float4*)gamma;
  const float4* b4 = (const float4*)beta;
#pragma unroll
  for (int j = 0; j < 8; ++j) {
    const int idx = j * 32 + lane;
    float4 g = g4[idx], b = b4[idx];
    float4 y;
    y.x = (d[j].x - mu) * rinv * g.x + b.x;
    y.y = (d[j].y - mu) * rinv * g.y + b.y;
    y.z = (d[j].z - mu) * rinv * g.z + b.z;
    y.w = (d[j].w - mu) * rinv * g.w + b.w;
    dstf[idx] = y;
    uint2 p;
    p.x = pack2bf(y.x, y.y);
    p.y = pack2bf(y.z, y.w);
    dstb[idx] = p;
  }
}

// ---------------------------------------------------------------------------
// Kernel 2: cast W [H,H] f32 -> bf16
// ---------------------------------------------------------------------------
__global__ void __launch_bounds__(256)
castw_kernel(const float* __restrict__ W, unsigned short* __restrict__ wbf) {
  const size_t i = (size_t)blockIdx.x * 256 + threadIdx.x;  // float4 index
  float4 f = ((const float4*)W)[i];
  uint2 p;
  p.x = pack2bf(f.x, f.y);
  p.y = pack2bf(f.z, f.w);
  ((uint2*)wbf)[i] = p;
}

// ---------------------------------------------------------------------------
// Fragment loaders for v_wmma_f32_16x16x32_bf16 (wave32).
// A (16x32): lanes 0-15 = rows 0-15 with K{0..15};
//            lanes 16-31 = rows 0-15 with K{16..31}.
// B (32x16): lane lc = column, lanes 0-15 K=k0..k0+15, lanes 16-31 K=+16.
// ---------------------------------------------------------------------------
__device__ __forceinline__ v16bf load_a(const unsigned short* arow, int k0,
                                        int half) {
  union { v16bf v; struct { v8bf lo, hi; } h; } a;
  a.h.lo = *(const v8bf*)(arow + k0 + half * 16);
  a.h.hi = *(const v8bf*)(arow + k0 + half * 16 + 8);
  return a.v;
}
__device__ __forceinline__ v16bf load_b(const unsigned short* brow, int k0) {
  union { v16bf v; struct { v8bf lo, hi; } h; } b;
  const v8bf* bp = (const v8bf*)(brow + k0);
  b.h.lo = bp[0];
  b.h.hi = bp[1];
  return b.v;
}

// ---------------------------------------------------------------------------
// Kernel 3: fused  scores[m] = clip( v . tanh(x[m]·W^T + bW) + bv )
// 256 threads = 8 waves = 4 m-tiles x 2 n-slots; 64 rows per workgroup.
// A from LDS-staged x tile (padded stride); B streamed from L2-resident W.
// K loop software-pipelined (prefetch depth 2) to overlap loads with WMMA.
// ---------------------------------------------------------------------------
__global__ void __launch_bounds__(256)
scores_kernel(const unsigned short* __restrict__ xbf,
              const unsigned short* __restrict__ wbf,
              const float* __restrict__ bW, const float* __restrict__ v,
              const float* __restrict__ bv, float* __restrict__ scores) {
  __shared__ float s_scores[64];
  extern __shared__ unsigned short xs[];   // 64 x LDS_STRIDE bf16

  const int tid = threadIdx.x;
  const int m0  = blockIdx.x * 64;

  // Stage the 64-row x tile into LDS with padded row stride.
  {
    const uint4* src = (const uint4*)(xbf + (size_t)m0 * HH);
    uint4*       dst = (uint4*)xs;
#pragma unroll
    for (int i = 0; i < 32; ++i) {
      const int e   = tid + i * 256;       // uint4 index, 8192 total
      const int row = e >> 7;              // /128 uint4 per source row
      const int col = e & 127;
      dst[row * (LDS_STRIDE / 8) + col] = src[e];
    }
  }
  if (tid < 64) s_scores[tid] = 0.0f;
  __syncthreads();

  const int wave = tid >> 5;
  const int lane = tid & 31;
  const int mt   = wave & 3;     // which 16-row m-tile
  const int ns   = wave >> 2;    // n-slot (even/odd 16-wide n-tiles)
  const int half = lane >> 4;    // lane group 0/1
  const int lc   = lane & 15;    // local row (A) / column (B)

  const unsigned short* arow = xs + (size_t)(mt * 16 + lc) * LDS_STRIDE;

  float acc[8];
#pragma unroll
  for (int r = 0; r < 8; ++r) acc[r] = 0.0f;

  for (int n0 = ns * 16; n0 < HH; n0 += 32) {
    const float vn = v[n0 + lc];     // issued early, consumed after K loop
    const float bn = bW[n0 + lc];
    const unsigned short* brow = wbf + (size_t)(n0 + lc) * HH + half * 16;

    v8f c = {0.f, 0.f, 0.f, 0.f, 0.f, 0.f, 0.f, 0.f};

    // Software pipeline: 2 fragments in flight ahead of consumption.
    v16bf a0 = load_a(arow, 0, half),  b0 = load_b(brow, 0);
    v16bf a1 = load_a(arow, 32, half), b1 = load_b(brow, 32);
#pragma unroll 2
    for (int k0 = 0; k0 + 64 < HH; k0 += 64) {
      v16bf a2 = load_a(arow, k0 + 64, half);
      v16bf b2 = load_b(brow, k0 + 64);
      c = __builtin_amdgcn_wmma_f32_16x16x32_bf16(false, a0, false, b0,
                                                  (short)0, c, false, false);
      v16bf a3 = load_a(arow, k0 + 96, half);
      v16bf b3 = load_b(brow, k0 + 96);
      c = __builtin_amdgcn_wmma_f32_16x16x32_bf16(false, a1, false, b1,
                                                  (short)0, c, false, false);
      a0 = a2; b0 = b2; a1 = a3; b1 = b3;
    }
    c = __builtin_amdgcn_wmma_f32_16x16x32_bf16(false, a0, false, b0,
                                                (short)0, c, false, false);
    c = __builtin_amdgcn_wmma_f32_16x16x32_bf16(false, a1, false, b1,
                                                (short)0, c, false, false);

#pragma unroll
    for (int r = 0; r < 8; ++r) acc[r] += TANHF(c[r] + bn) * vn;
  }

  // Reduce over the 16 N-lanes within each half, then across waves via LDS.
#pragma unroll
  for (int r = 0; r < 8; ++r) {
    float t = acc[r];
    t += __shfl_xor(t, 1, 32);
    t += __shfl_xor(t, 2, 32);
    t += __shfl_xor(t, 4, 32);
    t += __shfl_xor(t, 8, 32);
    if (lc == 0) atomicAdd(&s_scores[mt * 16 + half * 8 + r], t);
  }
  __syncthreads();

  if (tid < 64) {
    float s = s_scores[tid] + bv[0];
    if (s != s) s = 0.0f;                 // nan_to_num
    s = fminf(10.0f, fmaxf(-10.0f, s));   // clip (also catches +/-inf)
    scores[m0 + tid] = s;
  }
}

// ---------------------------------------------------------------------------
// Kernel 4: softmax over T=512 per batch row.
// ---------------------------------------------------------------------------
__global__ void __launch_bounds__(256)
softmax_kernel(const float* __restrict__ scores, float* __restrict__ weights) {
  __shared__ float redm[8];
  __shared__ float reds[8];
  const int b = blockIdx.x;
  const int tid = threadIdx.x;
  const float* s = scores + (size_t)b * TT;

  float a0 = s[tid], a1 = s[tid + 256];
  float m = fmaxf(a0, a1);
#pragma unroll
  for (int k = 1; k <= 16; k <<= 1) m = fmaxf(m, __shfl_xor(m, k, 32));
  if ((tid & 31) == 0) redm[tid >> 5] = m;
  __syncthreads();
  float gm = redm[0];
#pragma unroll
  for (int i = 1; i < 8; ++i) gm = fmaxf(gm, redm[i]);

  float e0 = __expf(a0 - gm);
  float e1 = __expf(a1 - gm);
  float ssum = e0 + e1;
#pragma unroll
  for (int k = 1; k <= 16; k <<= 1) ssum += __shfl_xor(ssum, k, 32);
  if ((tid & 31) == 0) reds[tid >> 5] = ssum;
  __syncthreads();
  float gs = 0.f;
#pragma unroll
  for (int i = 0; i < 8; ++i) gs += reds[i];

  const float inv = 1.0f / gs;
  float w0 = e0 * inv, w1 = e1 * inv;
  if (w0 != w0) w0 = 1.0f / TT;
  if (w1 != w1) w1 = 1.0f / TT;
  weights[(size_t)b * TT + tid]       = w0;
  weights[(size_t)b * TT + tid + 256] = w1;
}

// ---------------------------------------------------------------------------
// Kernel 5: context[b,h] = sum_t weights[b,t] * x_f32[b,t,h]
// Grid (BB, 8): block handles 128 h-values; 8 time-slices per block reduce
// through LDS. 512 blocks to keep HBM streaming wide.
// ---------------------------------------------------------------------------
__global__ void __launch_bounds__(256)
context_kernel(const float* __restrict__ xf, const float* __restrict__ weights,
               float* __restrict__ ctx) {
  __shared__ float wl[TT];
  __shared__ float4 part[256];
  const int b   = blockIdx.x;
  const int tid = threadIdx.x;
  wl[tid]       = weights[(size_t)b * TT + tid];
  wl[tid + 256] = weights[(size_t)b * TT + tid + 256];
  __syncthreads();

  const int lane32 = tid & 31;          // float4 index within 128-h chunk
  const int slice  = tid >> 5;          // 0..7 time slice
  const int h4     = blockIdx.y * 32 + lane32;

  const float4* x4 = (const float4*)(xf + (size_t)b * TT * HH);
  float4 acc = {0.f, 0.f, 0.f, 0.f};
  const int t0 = slice * (TT / 8);
#pragma unroll 4
  for (int t = t0; t < t0 + TT / 8; ++t) {
    const float w = wl[t];
    float4 xv = x4[(size_t)t * (HH / 4) + h4];
    acc.x += w * xv.x;
    acc.y += w * xv.y;
    acc.z += w * xv.z;
    acc.w += w * xv.w;
  }
  part[tid] = acc;
  __syncthreads();

  if (tid < 32) {
    float4 r = part[tid];
#pragma unroll
    for (int sl = 1; sl < 8; ++sl) {
      float4 p = part[tid + sl * 32];
      r.x += p.x; r.y += p.y; r.z += p.z; r.w += p.w;
    }
    ((float4*)(ctx + (size_t)b * HH))[blockIdx.y * 32 + tid] = r;
  }
}

// ---------------------------------------------------------------------------
extern "C" void kernel_launch(void* const* d_in, const int* in_sizes, int n_in,
                              void* d_out, int out_size, void* d_ws, size_t ws_size,
                              hipStream_t stream) {
  const float* lstm  = (const float*)d_in[0];  // [B,T,H] f32
  const float* W     = (const float*)d_in[1];  // [H,H]
  const float* b_W   = (const float*)d_in[2];  // [H]
  const float* v     = (const float*)d_in[3];  // [H]
  const float* b_v   = (const float*)d_in[4];  // scalar
  const float* gamma = (const float*)d_in[5];  // [H]
  const float* beta  = (const float*)d_in[6];  // [H]

  float* out     = (float*)d_out;
  float* context = out;                     // [B,H]
  float* weights = out + (size_t)BB * HH;   // [B,T]

  char* ws = (char*)d_ws;
  float*          xf     = (float*)ws;                                    // 128 MB
  unsigned short* xbf    = (unsigned short*)(ws + (size_t)ROWS * HH * 4); // 64 MB
  unsigned short* wbf    = xbf + (size_t)ROWS * HH;                       // 2 MB
  float*          scores = (float*)(wbf + (size_t)HH * HH);               // 128 KB

  ln_kernel<<<ROWS / 8, 256, 0, stream>>>(lstm, gamma, beta, xf, xbf);
  castw_kernel<<<(HH * HH) / 1024, 256, 0, stream>>>(W, wbf);
  scores_kernel<<<ROWS / 64, 256,
                  64 * LDS_STRIDE * sizeof(unsigned short), stream>>>(
      xbf, wbf, b_W, v, b_v, scores);
  softmax_kernel<<<BB, 256, 0, stream>>>(scores, weights);
  context_kernel<<<dim3(BB, 8), 256, 0, stream>>>(xf, weights, context);
}
